// Qwen3OmniMoeSparseMoeBlock_72971494359061
// MI455X (gfx1250) — matrix operand order repaired and checked by reading, not compiled, not run on hardware
//
#include <hip/hip_runtime.h>
#include <hip/hip_bf16.h>

// ---------------- problem constants ----------------
#define T_TOK 8192
#define D_DIM 2048
#define F_DIM 768
#define E_EXP 64
#define K_TOP 8
#define C_CAP 2048
#define TK (T_TOK * K_TOP)
#define GROUP_E 16

// ---------------- GEMM tiling ----------------
#define BM 256
#define BN 128
#define BK 32
#define LDW 18   // dwords per LDS row (16 data + 2 pad; even -> 8B-aligned rows)

typedef __attribute__((ext_vector_type(16))) __bf16 v16bf;
typedef __attribute__((ext_vector_type(8)))  float  v8f;
typedef __attribute__((ext_vector_type(8)))  unsigned int v8u;

__device__ __forceinline__ unsigned int pack2_bf16(float a, float b) {
  unsigned int ua = __float_as_uint(a);
  ua = (ua + 0x7FFFu + ((ua >> 16) & 1u)) >> 16;
  unsigned int ub = __float_as_uint(b);
  ub = (ub + 0x7FFFu + ((ub >> 16) & 1u)) >> 16;
  return ua | (ub << 16);
}
__device__ __forceinline__ unsigned short f2bf(float a) {
  unsigned int ua = __float_as_uint(a);
  return (unsigned short)((ua + 0x7FFFu + ((ua >> 16) & 1u)) >> 16);
}

// A-matrix 16x32 bf16 frag (ISA 7.12.2): lane m = lane&15,
// vgpr v half h: k = (v&3)*2 + h + 8*(lane>=16) + 16*(v>=4)
__device__ __forceinline__ v16bf load_a_frag(const unsigned int* __restrict__ s,
                                             int row, int lane) {
  v8u u;
  const unsigned int* p = s + row * LDW;
  const int kb = (lane >> 4) << 2;
#pragma unroll
  for (int v = 0; v < 8; ++v)
    u[v] = p[(v & 3) + kb + ((v >> 2) << 3)];
  return __builtin_bit_cast(v16bf, u);
}
// B-matrix 32x16 bf16 frag: lane n = lane&15, vgpr v half h:
// k = (v&3)*2 + h + 8*(lane>=16) + 16*(v>=4). LDS stores B transposed: [n][k/2] dwords.
__device__ __forceinline__ v16bf load_b_frag(const unsigned int* __restrict__ s,
                                             int col, int lane) {
  v8u u;
  const unsigned int* p = s + col * LDW;
  const int kb = (lane >> 4) << 2;
#pragma unroll
  for (int v = 0; v < 8; ++v)
    u[v] = p[(v & 3) + kb + ((v >> 2) << 3)];
  return __builtin_bit_cast(v16bf, u);
}

// ---------------- kernel 0: init out / tok / wbuf ----------------
__global__ void init_kernel(float* __restrict__ out, int* __restrict__ tok,
                            float* __restrict__ wbuf) {
  size_t i = (size_t)blockIdx.x * blockDim.x + threadIdx.x;
  size_t stride = (size_t)gridDim.x * blockDim.x;
  for (size_t k = i; k < (size_t)T_TOK * D_DIM; k += stride) out[k] = 0.0f;
  for (size_t k = i; k < (size_t)E_EXP * C_CAP; k += stride) {
    tok[k] = T_TOK;
    wbuf[k] = 0.0f;
  }
}

// ---------------- kernel 1: router + softmax + top-8 ----------------
__global__ __launch_bounds__(64) void router_topk_kernel(
    const float* __restrict__ x, const float* __restrict__ wr,
    int* __restrict__ topi, float* __restrict__ topw) {
  __shared__ float xs[64];
  __shared__ float sl[64];
  __shared__ float rv[64];
  __shared__ int   ri[64];

  const int t = blockIdx.x;
  const int tid = threadIdx.x;
  const float* xr = x + (size_t)t * D_DIM;

  float acc = 0.0f;
  for (int d0 = 0; d0 < D_DIM; d0 += 64) {
    xs[tid] = xr[d0 + tid];
    __syncthreads();
#pragma unroll 16
    for (int dd = 0; dd < 64; ++dd)
      acc += xs[dd] * wr[(size_t)(d0 + dd) * E_EXP + tid];
    __syncthreads();
  }

  // softmax over 64 logits
  rv[tid] = acc;
  __syncthreads();
  for (int off = 32; off > 0; off >>= 1) {
    if (tid < off) rv[tid] = fmaxf(rv[tid], rv[tid + off]);
    __syncthreads();
  }
  const float mx = rv[0];
  __syncthreads();
  float p = __expf(acc - mx);
  rv[tid] = p;
  __syncthreads();
  for (int off = 32; off > 0; off >>= 1) {
    if (tid < off) rv[tid] += rv[tid + off];
    __syncthreads();
  }
  const float sm = rv[0];
  __syncthreads();
  p /= sm;
  sl[tid] = p;
  __syncthreads();

  // iterative top-8 argmax (strict '>' keeps lowest index on ties)
  float wsel[K_TOP];
  int isel[K_TOP];
#pragma unroll
  for (int k = 0; k < K_TOP; ++k) {
    rv[tid] = sl[tid];
    ri[tid] = tid;
    __syncthreads();
    for (int off = 32; off > 0; off >>= 1) {
      if (tid < off && rv[tid + off] > rv[tid]) {
        rv[tid] = rv[tid + off];
        ri[tid] = ri[tid + off];
      }
      __syncthreads();
    }
    wsel[k] = rv[0];
    isel[k] = ri[0];
    __syncthreads();
    if (tid == ri[0]) sl[tid] = -1.0f;
    __syncthreads();
  }
  float tot = 0.0f;
#pragma unroll
  for (int k = 0; k < K_TOP; ++k) tot += wsel[k];
  if (tid < K_TOP) {
    topi[(size_t)t * K_TOP + tid] = isel[tid];
    topw[(size_t)t * K_TOP + tid] = wsel[tid] / tot;
  }
}

// ---------------- kernel 2: deterministic routing (stable per-expert scan) ---
__global__ __launch_bounds__(256) void route_kernel(
    const int* __restrict__ topi, const float* __restrict__ topw,
    int* __restrict__ tok, float* __restrict__ wbuf) {
  __shared__ int s[256];
  __shared__ int sbase;
  const int e = blockIdx.x;
  const int tid = threadIdx.x;
  if (tid == 0) sbase = 0;
  __syncthreads();
  for (int i0 = 0; i0 < TK; i0 += 256) {
    const int idx = i0 + tid;
    const int match = (topi[idx] == e) ? 1 : 0;
    s[tid] = match;
    __syncthreads();
    for (int off = 1; off < 256; off <<= 1) {
      int tmp = (tid >= off) ? s[tid - off] : 0;
      __syncthreads();
      s[tid] += tmp;
      __syncthreads();
    }
    const int pos = sbase + s[tid] - match;
    const int total = s[255];
    if (match && pos < C_CAP) {
      tok[e * C_CAP + pos] = idx >> 3;  // token = flat/K
      wbuf[e * C_CAP + pos] = topw[idx];
    }
    __syncthreads();
    if (tid == 0) sbase += total;
    __syncthreads();
  }
}

// ---------------- kernel 3: gate+up GEMM + SiLU ----------------
// BM=256 x BN=128 block tile; 8 waves in 4(m) x 2(n) grid; 64x64 per wave.
__global__ __launch_bounds__(256, 1) void gateup_kernel(
    const float* __restrict__ x, const float* __restrict__ w_gate,
    const float* __restrict__ w_up, const int* __restrict__ tok,
    unsigned short* __restrict__ H, int e0) {
  __shared__ unsigned int As[BM * LDW];
  __shared__ unsigned int Bg[BN * LDW];
  __shared__ unsigned int Bu[BN * LDW];
  __shared__ int tokRow[BM];

  const int tid = threadIdx.x;
  const int lane = tid & 31;
  const int wid = tid >> 5;
  const int wm = wid & 3;    // 4 wave rows of 64
  const int wn = wid >> 2;   // 2 wave cols of 64
  const int nbase = blockIdx.x * BN;
  const int mbase = blockIdx.y * BM;
  const int eL = blockIdx.z;
  const int e = e0 + eL;

  tokRow[tid] = tok[e * C_CAP + mbase + tid];
  __syncthreads();

  v8f accG[4][4], accU[4][4];
#pragma unroll
  for (int i = 0; i < 4; ++i)
#pragma unroll
    for (int j = 0; j < 4; ++j) {
      accG[i][j] = (v8f)0.0f;
      accU[i][j] = (v8f)0.0f;
    }

  // B loader mapping: thread covers 2 f-cols and 2 d-rows per pass (packed k)
  const int bf0 = (tid & 63) * 2;        // 0..126, pairs of f columns
  const int bdb = (tid >> 6) * 2;        // 0,2,4,6
  const int myTok = tokRow[tid];         // A: one row per thread

  for (int kk = 0; kk < D_DIM; kk += BK) {
    // --- gather A (tokens) -> bf16 LDS, one full row (32 floats) / thread ---
    {
      unsigned int* dst = As + tid * LDW;
      if (myTok < T_TOK) {
        const float* src = x + (size_t)myTok * D_DIM + kk;
#pragma unroll
        for (int q = 0; q < 4; ++q) {
          float4 f0 = *(const float4*)(src + q * 8);
          float4 f1 = *(const float4*)(src + q * 8 + 4);
          uint2 u;
          u.x = pack2_bf16(f0.x, f0.y);
          u.y = pack2_bf16(f0.z, f0.w);
          *(uint2*)(dst + q * 4) = u;
          u.x = pack2_bf16(f1.x, f1.y);
          u.y = pack2_bf16(f1.z, f1.w);
          *(uint2*)(dst + q * 4 + 2) = u;
        }
      } else {
        uint2 z; z.x = 0u; z.y = 0u;
#pragma unroll
        for (int q = 0; q < 8; ++q) *(uint2*)(dst + q * 2) = z;
      }
    }
    // --- B tiles (gate & up): load 2 k-rows, pack dwords, ds_store_b32 ---
#pragma unroll
    for (int pass = 0; pass < 4; ++pass) {
      const int dl = bdb + pass * 8;  // even k row
      const size_t gi0 = ((size_t)e * D_DIM + kk + dl) * F_DIM + nbase + bf0;
      const size_t gi1 = gi0 + F_DIM;  // next k row
      float2 g0 = *(const float2*)(w_gate + gi0);
      float2 g1 = *(const float2*)(w_gate + gi1);
      float2 u0 = *(const float2*)(w_up + gi0);
      float2 u1 = *(const float2*)(w_up + gi1);
      const int kd = dl >> 1;
      Bg[(bf0 + 0) * LDW + kd] = pack2_bf16(g0.x, g1.x);
      Bg[(bf0 + 1) * LDW + kd] = pack2_bf16(g0.y, g1.y);
      Bu[(bf0 + 0) * LDW + kd] = pack2_bf16(u0.x, u1.x);
      Bu[(bf0 + 1) * LDW + kd] = pack2_bf16(u0.y, u1.y);
    }
    __syncthreads();

    v16bf a[4], bg[4], bu[4];
#pragma unroll
    for (int i = 0; i < 4; ++i)
      a[i] = load_a_frag(As, wm * 64 + i * 16 + (lane & 15), lane);
#pragma unroll
    for (int j = 0; j < 4; ++j) {
      bg[j] = load_b_frag(Bg, wn * 64 + j * 16 + (lane & 15), lane);
      bu[j] = load_b_frag(Bu, wn * 64 + j * 16 + (lane & 15), lane);
    }
#pragma unroll
    for (int i = 0; i < 4; ++i)
#pragma unroll
      for (int j = 0; j < 4; ++j) {
        accG[i][j] = __builtin_amdgcn_wmma_f32_16x16x32_bf16(
            false, a[i], false, bg[j], (short)0, accG[i][j], false, false);
        accU[i][j] = __builtin_amdgcn_wmma_f32_16x16x32_bf16(
            false, a[i], false, bu[j], (short)0, accU[i][j], false, false);
      }
    __syncthreads();
  }

  // --- epilogue: h = silu(g)*u -> bf16 H[eL][slot][f] ---
  const size_t hbase = (size_t)eL * C_CAP * F_DIM;
#pragma unroll
  for (int i = 0; i < 4; ++i)
#pragma unroll
    for (int j = 0; j < 4; ++j) {
      const int c = nbase + wn * 64 + j * 16 + (lane & 15);
      const int rb = mbase + wm * 64 + i * 16 + 8 * (lane >> 4);
#pragma unroll
      for (int v = 0; v < 8; ++v) {
        const float g = accG[i][j][v];
        const float h = (g / (1.0f + __expf(-g))) * accU[i][j][v];
        H[hbase + (size_t)(rb + v) * F_DIM + c] = f2bf(h);
      }
    }
}

// ---------------- kernel 4: down GEMM + weight + scatter-add ----------------
__global__ __launch_bounds__(256, 1) void down_kernel(
    const unsigned short* __restrict__ H, const float* __restrict__ w_down,
    const int* __restrict__ tok, const float* __restrict__ wbuf,
    float* __restrict__ out, int e0) {
  __shared__ unsigned int As[BM * LDW];
  __shared__ unsigned int Bd[BN * LDW];
  __shared__ int tokRow[BM];
  __shared__ float wRow[BM];

  const int tid = threadIdx.x;
  const int lane = tid & 31;
  const int wid = tid >> 5;
  const int wm = wid & 3;
  const int wn = wid >> 2;
  const int nbase = blockIdx.x * BN;  // D columns
  const int mbase = blockIdx.y * BM;
  const int eL = blockIdx.z;
  const int e = e0 + eL;

  tokRow[tid] = tok[e * C_CAP + mbase + tid];
  wRow[tid] = wbuf[e * C_CAP + mbase + tid];
  __syncthreads();

  v8f acc[4][4];
#pragma unroll
  for (int i = 0; i < 4; ++i)
#pragma unroll
    for (int j = 0; j < 4; ++j) acc[i][j] = (v8f)0.0f;

  const int bf0 = (tid & 63) * 2;
  const int bdb = (tid >> 6) * 2;
  const unsigned short* const hrow =
      H + ((size_t)eL * C_CAP + mbase + tid) * F_DIM;

  for (int kk = 0; kk < F_DIM; kk += BK) {
    // --- A tile from H (already bf16), one full row / thread ---
    {
      const unsigned short* hp = hrow + kk;
      unsigned int* dst = As + tid * LDW;
#pragma unroll
      for (int q = 0; q < 4; ++q) {
        const uint2 u = *(const uint2*)(hp + q * 8);
        *(uint2*)(dst + q * 4) = u;
        const uint2 u2 = *(const uint2*)(hp + q * 8 + 4);
        *(uint2*)(dst + q * 4 + 2) = u2;
      }
    }
    // --- B tile from w_down: load 2 k-rows, pack dwords ---
#pragma unroll
    for (int pass = 0; pass < 4; ++pass) {
      const int fl = bdb + pass * 8;
      const size_t gi0 = ((size_t)e * F_DIM + kk + fl) * D_DIM + nbase + bf0;
      const size_t gi1 = gi0 + D_DIM;
      float2 d0 = *(const float2*)(w_down + gi0);
      float2 d1 = *(const float2*)(w_down + gi1);
      const int kd = fl >> 1;
      Bd[(bf0 + 0) * LDW + kd] = pack2_bf16(d0.x, d1.x);
      Bd[(bf0 + 1) * LDW + kd] = pack2_bf16(d0.y, d1.y);
    }
    __syncthreads();

    v16bf a[4], b[4];
#pragma unroll
    for (int i = 0; i < 4; ++i)
      a[i] = load_a_frag(As, wm * 64 + i * 16 + (lane & 15), lane);
#pragma unroll
    for (int j = 0; j < 4; ++j)
      b[j] = load_b_frag(Bd, wn * 64 + j * 16 + (lane & 15), lane);
#pragma unroll
    for (int i = 0; i < 4; ++i)
#pragma unroll
      for (int j = 0; j < 4; ++j)
        acc[i][j] = __builtin_amdgcn_wmma_f32_16x16x32_bf16(
            false, a[i], false, b[j], (short)0, acc[i][j], false, false);
    __syncthreads();
  }

  // --- epilogue: y = acc * w_slot, scatter-add into out[token] ---
#pragma unroll
  for (int i = 0; i < 4; ++i)
#pragma unroll
    for (int j = 0; j < 4; ++j) {
      const int c = nbase + wn * 64 + j * 16 + (lane & 15);
      const int r0 = wm * 64 + i * 16 + 8 * (lane >> 4);
#pragma unroll
      for (int v = 0; v < 8; ++v) {
        const int r = r0 + v;
        const int tokv = tokRow[r];
        if (tokv < T_TOK) {
          const float y = acc[i][j][v] * wRow[r];
          atomicAdd(out + (size_t)tokv * D_DIM + c, y);
        }
      }
    }
}

// ---------------- host launcher ----------------
extern "C" void kernel_launch(void* const* d_in, const int* in_sizes, int n_in,
                              void* d_out, int out_size, void* d_ws,
                              size_t ws_size, hipStream_t stream) {
  const float* x = (const float*)d_in[0];
  const float* w_router = (const float*)d_in[1];
  const float* w_gate = (const float*)d_in[2];
  const float* w_up = (const float*)d_in[3];
  const float* w_down = (const float*)d_in[4];
  float* out = (float*)d_out;

  char* ws = (char*)d_ws;
  size_t off = 0;
  int* topi = (int*)(ws + off);          off += (size_t)TK * sizeof(int);
  float* topw = (float*)(ws + off);      off += (size_t)TK * sizeof(float);
  int* tok = (int*)(ws + off);           off += (size_t)E_EXP * C_CAP * sizeof(int);
  float* wbuf = (float*)(ws + off);      off += (size_t)E_EXP * C_CAP * sizeof(float);
  unsigned short* H = (unsigned short*)(ws + off);
  // H: GROUP_E * C * F bf16 = 48 MB, reused per expert group

  init_kernel<<<4096, 256, 0, stream>>>(out, tok, wbuf);
  router_topk_kernel<<<T_TOK, 64, 0, stream>>>(x, w_router, topi, topw);
  route_kernel<<<E_EXP, 256, 0, stream>>>(topi, topw, tok, wbuf);

  for (int e0 = 0; e0 < E_EXP; e0 += GROUP_E) {
    dim3 ga(F_DIM / BN, C_CAP / BM, GROUP_E);  // (6,8,16)
    gateup_kernel<<<ga, 256, 0, stream>>>(x, w_gate, w_up, tok, H, e0);
    dim3 gb(D_DIM / BN, C_CAP / BM, GROUP_E);  // (16,8,16)
    down_kernel<<<gb, 256, 0, stream>>>(H, w_down, tok, wbuf, out, e0);
  }
}
